// ExpansionContrastModule_52381421142697
// MI455X (gfx1250) — matrix operand order repaired and checked by reading, not compiled, not run on hardware
//
#include <hip/hip_runtime.h>

typedef __bf16 bf16_t;
typedef __attribute__((ext_vector_type(16))) __bf16 v16bf;
typedef __attribute__((ext_vector_type(8)))  __bf16 v8bf;
typedef __attribute__((ext_vector_type(8)))  float  v8f;
typedef __attribute__((ext_vector_type(4)))  int    v4i;

#define B_N 4
#define C_N 256
#define H_N 96
#define W_N 96
#define G_N 9
#define TILE_W 32
#define HALO_W 34
#define HALO_PIX (3 * HALO_W)   // 102 pixels (3 rows x 34 cols)
#define CH_STR 264              // padded channel stride in LDS (16B-aligned rows, bank spread)

#define HAS_ASYNC_LDS __has_builtin(__builtin_amdgcn_global_load_async_to_lds_b128)

__device__ __forceinline__ v16bf cat8(v8bf lo, v8bf hi) {
  return __builtin_shufflevector(lo, hi, 0,1,2,3,4,5,6,7,8,9,10,11,12,13,14,15);
}

// Pre-pass 1: W3' = bf16( g<8 ? -W3[g] : W3[g] )  (folds the surround negation)
__global__ void w3_prep_kernel(const float* __restrict__ w3, bf16_t* __restrict__ w3n) {
  int i = blockIdx.x * 256 + threadIdx.x;
  if (i < G_N * C_N * C_N) {
    float v = w3[i];
    if ((i >> 16) < 8) v = -v;      // C_N*C_N == 65536
    w3n[i] = (bf16_t)v;
  }
}

// Pre-pass 2: transpose cen (B,C,H,W fp32) -> cen_t (B,H,W,C bf16), coalesced both sides via LDS.
__global__ void cen_tr_kernel(const float* __restrict__ cen, bf16_t* __restrict__ cen_t) {
  __shared__ float tile[64][W_N + 1];
  const int blk = blockIdx.x;                 // (b, y, c-block of 64)
  const int cb = blk % (C_N / 64);
  const int y  = (blk / (C_N / 64)) % H_N;
  const int b  =  blk / ((C_N / 64) * H_N);
  const int c0 = cb * 64;

  const float* __restrict__ src = cen + ((size_t)(b * C_N + c0) * H_N + y) * W_N;
  for (int idx = threadIdx.x; idx < 64 * W_N; idx += 256) {
    int ci = idx / W_N, x = idx - ci * W_N;   // consecutive tid -> consecutive x (coalesced)
    tile[ci][x] = src[(size_t)ci * (H_N * W_N) + x];
  }
  __syncthreads();
  bf16_t* __restrict__ dst = cen_t + ((size_t)(b * H_N + y) * W_N) * C_N + c0;
  for (int idx = threadIdx.x; idx < W_N * 64; idx += 256) {
    int x = idx >> 6, ci = idx & 63;          // consecutive tid -> consecutive c (coalesced)
    dst[(size_t)x * C_N + ci] = (bf16_t)tile[ci][x];
  }
}

template <bool TRANSPOSED>
__launch_bounds__(256, 2)
__global__ void ecm_fused_kernel(const float* __restrict__ cen,
                                 const bf16_t* __restrict__ cen_t,
                                 const bf16_t* __restrict__ w3n,
                                 float* __restrict__ out) {
  __shared__ __attribute__((aligned(16))) bf16_t s_tile[HALO_PIX * CH_STR];
  __shared__ float s_norm2[G_N][2][16];

  const int tid  = threadIdx.x;
  const int wave = tid >> 5;
  const int lane = tid & 31;
  const int l16  = lane & 15;
  const bool hi  = lane >= 16;

  const int blk = blockIdx.x;
  const int wt  = blk % (W_N / TILE_W);
  const int h   = (blk / (W_N / TILE_W)) % H_N;
  const int b   =  blk / ((W_N / TILE_W) * H_N);
  const int w0  = wt * TILE_W;

  for (int i = tid; i < G_N * 2 * 16; i += 256) (&s_norm2[0][0][0])[i] = 0.0f;

  const float* __restrict__ cen_b = cen + (size_t)b * (C_N * H_N * W_N);

  if constexpr (TRANSPOSED) {
    // Stage halo tile from pixel-major bf16 with async global->LDS copies (no VGPR round-trip).
    const bf16_t* __restrict__ ct_b = cen_t + (size_t)b * (H_N * W_N * C_N);
    for (int idx = tid; idx < HALO_PIX * 32; idx += 256) {   // 32 x 16B chunks per pixel
      int p  = idx >> 5;
      int q  = idx & 31;
      int hp = p / HALO_W;
      int cp = p - hp * HALO_W;
      int y  = h  - 1 + hp;
      int x  = w0 - 1 + cp;
      bf16_t* ldst = s_tile + p * CH_STR + q * 8;
      if ((unsigned)y < (unsigned)H_N && (unsigned)x < (unsigned)W_N) {
        const bf16_t* g = ct_b + ((size_t)(y * W_N + x)) * C_N + q * 8;
#if HAS_ASYNC_LDS
        __builtin_amdgcn_global_load_async_to_lds_b128(
            (__attribute__((address_space(1))) v4i*)(void*)(const void*)g,
            (__attribute__((address_space(3))) v4i*)(void*)ldst, 0, 0);
#else
        *(v8bf*)ldst = *(const v8bf*)g;
#endif
      } else {
        v8bf z = {};
        *(v8bf*)ldst = z;                                    // zero padding at borders
      }
    }
#if HAS_ASYNC_LDS
#if __has_builtin(__builtin_amdgcn_s_wait_asynccnt)
    __builtin_amdgcn_s_wait_asynccnt(0);
#else
    asm volatile("s_wait_asynccnt 0x0" ::: "memory");
#endif
#endif
  } else {
    // Fallback: stage from channel-major fp32 with on-the-fly bf16 conversion.
    for (int idx = tid; idx < HALO_PIX * C_N; idx += 256) {
      int ch = idx / HALO_PIX;
      int p  = idx - ch * HALO_PIX;
      int hp = p / HALO_W;
      int cp = p - hp * HALO_W;
      int y  = h  - 1 + hp;
      int x  = w0 - 1 + cp;
      float v = 0.0f;
      if ((unsigned)y < (unsigned)H_N && (unsigned)x < (unsigned)W_N)
        v = cen_b[ch * (H_N * W_N) + y * W_N + x];
      s_tile[p * CH_STR + ch] = (bf16_t)v;
    }
  }
  __syncthreads();

  // Each wave owns M-tiles {2w, 2w+1} x N-tiles {0,1}
  v8f o00 = {}, o01 = {}, o10 = {}, o11 = {};
  const int m0    = wave * 2;
  const int rowA0 = (m0 * 16 + l16) * C_N;
  const int rowA1 = rowA0 + 16 * C_N;

  // halo coords of the source pixel for each group (offset+1)
  const int ry_t[G_N] = {0,0,0,1,2,2,2,1,1};
  const int cx_t[G_N] = {0,1,2,2,2,1,0,0,1};

  #pragma unroll 1
  for (int g = 0; g < G_N; ++g) {
    const bf16_t* __restrict__ Ag = w3n + g * (C_N * C_N);
    const int pix0 = (ry_t[g] * HALO_W + cx_t[g] + l16) * CH_STR;
    const int pix1 = pix0 + 16 * CH_STR;
    const int ka = hi ? 8 : 0;     // A: lanes 16-31 carry K+8 / K+24 chunks
    const int kb = hi ? 8 : 0;     // B: lanes 16-31 carry K+8 / K+24 chunks

    v8f c00 = {}, c01 = {}, c10 = {}, c11 = {};
    #pragma unroll
    for (int k0 = 0; k0 < C_N; k0 += 32) {
      v16bf A0 = cat8(*(const v8bf*)(Ag + rowA0 + k0 + ka),
                      *(const v8bf*)(Ag + rowA0 + k0 + ka + 16));
      v16bf A1 = cat8(*(const v8bf*)(Ag + rowA1 + k0 + ka),
                      *(const v8bf*)(Ag + rowA1 + k0 + ka + 16));
      v16bf B0 = cat8(*(const v8bf*)(s_tile + pix0 + k0 + kb),
                      *(const v8bf*)(s_tile + pix0 + k0 + kb + 16));
      v16bf B1 = cat8(*(const v8bf*)(s_tile + pix1 + k0 + kb),
                      *(const v8bf*)(s_tile + pix1 + k0 + kb + 16));
      c00 = __builtin_amdgcn_wmma_f32_16x16x32_bf16(false, A0, false, B0, (short)0, c00, false, false);
      c01 = __builtin_amdgcn_wmma_f32_16x16x32_bf16(false, A0, false, B1, (short)0, c01, false, false);
      c10 = __builtin_amdgcn_wmma_f32_16x16x32_bf16(false, A1, false, B0, (short)0, c10, false, false);
      c11 = __builtin_amdgcn_wmma_f32_16x16x32_bf16(false, A1, false, B1, (short)0, c11, false, false);
    }

    // per-column sum of squares over this wave's 32 output channels
    float t0 = 0.0f, t1 = 0.0f;
    #pragma unroll
    for (int r = 0; r < 8; ++r) {
      t0 += c00[r] * c00[r] + c10[r] * c10[r];
      t1 += c01[r] * c01[r] + c11[r] * c11[r];
    }
    t0 += __shfl_xor(t0, 16, 32);   // combine M halves (lanes l <-> l+16 share a column)
    t1 += __shfl_xor(t1, 16, 32);
    if (lane < 16) {
      atomicAdd(&s_norm2[g][0][l16], t0);
      atomicAdd(&s_norm2[g][1][l16], t1);
    }
    __syncthreads();

    const float inv0 = 1.0f / fmaxf(sqrtf(s_norm2[g][0][l16]), 1e-12f);
    const float inv1 = 1.0f / fmaxf(sqrtf(s_norm2[g][1][l16]), 1e-12f);
    #pragma unroll
    for (int r = 0; r < 8; ++r) {
      o00[r] += c00[r] * inv0;  o10[r] += c10[r] * inv0;
      o01[r] += c01[r] * inv1;  o11[r] += c11[r] * inv1;
    }
  }

  // Epilogue: out = Sigma_g normalized + cen (fp32)
  float* __restrict__ out_b = out + (size_t)b * (C_N * H_N * W_N);
  const int x0 = w0 + l16, x1 = x0 + 16;
  const int rbase = h * W_N;
  #pragma unroll
  for (int r = 0; r < 8; ++r) {
    int i0 = m0 * 16 + r + (hi ? 8 : 0);
    int i1 = i0 + 16;
    size_t a0 = (size_t)i0 * (H_N * W_N) + rbase;
    size_t a1 = (size_t)i1 * (H_N * W_N) + rbase;
    out_b[a0 + x0] = o00[r] + cen_b[a0 + x0];
    out_b[a0 + x1] = o01[r] + cen_b[a0 + x1];
    out_b[a1 + x0] = o10[r] + cen_b[a1 + x0];
    out_b[a1 + x1] = o11[r] + cen_b[a1 + x1];
  }
}

extern "C" void kernel_launch(void* const* d_in, const int* in_sizes, int n_in,
                              void* d_out, int out_size, void* d_ws, size_t ws_size,
                              hipStream_t stream) {
  const float* cen = (const float*)d_in[0];
  const float* W3  = (const float*)d_in[3];   // W1/W2 mathematically dead (softmax over size-1 axis == 1)

  bf16_t* w3n = (bf16_t*)d_ws;                               // 9*256*256 bf16 = 1.18 MB
  const size_t w3_bytes   = (size_t)G_N * C_N * C_N * sizeof(bf16_t);
  const size_t cent_bytes = (size_t)B_N * H_N * W_N * C_N * sizeof(bf16_t);  // 18.9 MB

  w3_prep_kernel<<<(G_N * C_N * C_N + 255) / 256, 256, 0, stream>>>(W3, w3n);

  const int grid = B_N * H_N * (W_N / TILE_W);
  if (ws_size >= w3_bytes + cent_bytes) {
    bf16_t* cen_t = (bf16_t*)((char*)d_ws + w3_bytes);
    cen_tr_kernel<<<B_N * H_N * (C_N / 64), 256, 0, stream>>>(cen, cen_t);
    ecm_fused_kernel<true><<<grid, 256, 0, stream>>>(cen, cen_t, w3n, (float*)d_out);
  } else {
    ecm_fused_kernel<false><<<grid, 256, 0, stream>>>(cen, nullptr, w3n, (float*)d_out);
  }
}
